// SRGL_Model_26096221290700
// MI455X (gfx1250) — matrix operand order repaired and checked by reading, not compiled, not run on hardware
//
#include <hip/hip_runtime.h>
#include <math.h>

typedef __attribute__((ext_vector_type(16))) __bf16 v16bf;
typedef __attribute__((ext_vector_type(8)))  __bf16 v8bf;
typedef __attribute__((ext_vector_type(4)))  __bf16 v4bf;
typedef __attribute__((ext_vector_type(8)))  float  v8f;
typedef __attribute__((ext_vector_type(4)))  float  v4f;

#define D_NUM 4000
#define T_NUM 2000
#define FEAT  256
#define UNITS 200
#define TOPK  10

#define BM 64
#define BN 64
#define BK 32
#define BKP 40   // padded LDS stride in bf16 elems: 80 bytes (rows 8B/16B aligned)

// ---------------------------------------------------------------------------
// Block-tiled bf16 WMMA GEMM: C = act(rowScale[m] * (opA(A) @ opB(B)))
// opA(A) is M x K (TA: A stored K x M), opB(B) is K x N (TB: B stored N x K).
// 128 threads = 4 waves; block computes a 64x64 C tile; each wave a 16x64
// strip = one A fragment x four B fragments per 32-wide k-step
// (4 x v_wmma_f32_16x16x32_bf16 per iteration).
// Staging: float4 global loads in the coalesced direction (all dims and
// leading dims are multiples of 4; tile origins multiples of 64, so vector
// loads are aligned and bounds are all-or-nothing), 32-bit offset math,
// layout specialization via template -> no per-iteration layout branching.
// act: 0 = none, 1 = relu, 2 = sigmoid
// ---------------------------------------------------------------------------
template <int TA, int TB>
__global__ __launch_bounds__(128)
void gemm_wmma_kernel(const float* __restrict__ A, const float* __restrict__ B,
                      float* __restrict__ C, int M, int N, int K,
                      int lda, int ldb, int ldc,
                      const float* __restrict__ rowScale, int act)
{
    __shared__ __align__(16) __bf16 As[BM * BKP];   // As[m][k]
    __shared__ __align__(16) __bf16 Bs[BN * BKP];   // Bs[n][k]  (k-contiguous)

    const int m0   = blockIdx.y * BM;
    const int n0   = blockIdx.x * BN;
    const int tid  = threadIdx.x;
    const int wave = tid >> 5;
    const int lane = tid & 31;
    const int half = lane >> 4;
    const int l    = lane & 15;

    const v4f vzero = {0.0f, 0.0f, 0.0f, 0.0f};
    v8f acc[4] = {};

    for (int k0 = 0; k0 < K; k0 += BK) {
        // ---- stage A tile (BM x BK) as bf16 ----
        if (TA == 0) {
            // A row-major M x K: coalesced float4 along k
#pragma unroll
            for (int i = 0; i < 4; ++i) {
                int li = tid + i * 128;
                int k4 = (li & 7) * 4;          // 8 float4 per 32-wide row
                int m  = li >> 3;
                int gm = m0 + m, gk = k0 + k4;
                bool ok = (gm < M) && (gk < K);
                int off = ok ? (gm * lda + gk) : 0;
                v4f v = *(const v4f*)(A + off);
                v = ok ? v : vzero;
                v4bf w;
                w[0] = (__bf16)v[0]; w[1] = (__bf16)v[1];
                w[2] = (__bf16)v[2]; w[3] = (__bf16)v[3];
                *(v4bf*)&As[m * BKP + k4] = w;
            }
        } else {
            // A stored K x M: coalesced float4 along m, scatter into As
#pragma unroll
            for (int i = 0; i < 4; ++i) {
                int li = tid + i * 128;
                int m4 = (li & 15) * 4;         // 16 float4 per 64-wide row
                int k  = li >> 4;
                int gm = m0 + m4, gk = k0 + k;
                bool ok = (gm < M) && (gk < K);
                int off = ok ? (gk * lda + gm) : 0;
                v4f v = *(const v4f*)(A + off);
                v = ok ? v : vzero;
                As[(m4 + 0) * BKP + k] = (__bf16)v[0];
                As[(m4 + 1) * BKP + k] = (__bf16)v[1];
                As[(m4 + 2) * BKP + k] = (__bf16)v[2];
                As[(m4 + 3) * BKP + k] = (__bf16)v[3];
            }
        }
        // ---- stage B tile transposed into Bs[n][k] ----
        if (TB == 0) {
            // B row-major K x N: coalesced float4 along n, scatter into Bs
#pragma unroll
            for (int i = 0; i < 4; ++i) {
                int li = tid + i * 128;
                int n4 = (li & 15) * 4;
                int k  = li >> 4;
                int gn = n0 + n4, gk = k0 + k;
                bool ok = (gn < N) && (gk < K);
                int off = ok ? (gk * ldb + gn) : 0;
                v4f v = *(const v4f*)(B + off);
                v = ok ? v : vzero;
                Bs[(n4 + 0) * BKP + k] = (__bf16)v[0];
                Bs[(n4 + 1) * BKP + k] = (__bf16)v[1];
                Bs[(n4 + 2) * BKP + k] = (__bf16)v[2];
                Bs[(n4 + 3) * BKP + k] = (__bf16)v[3];
            }
        } else {
            // B stored N x K: coalesced float4 along k, contiguous into Bs
#pragma unroll
            for (int i = 0; i < 4; ++i) {
                int li = tid + i * 128;
                int k4 = (li & 7) * 4;
                int n  = li >> 3;
                int gn = n0 + n, gk = k0 + k4;
                bool ok = (gn < N) && (gk < K);
                int off = ok ? (gn * ldb + gk) : 0;
                v4f v = *(const v4f*)(B + off);
                v = ok ? v : vzero;
                v4bf w;
                w[0] = (__bf16)v[0]; w[1] = (__bf16)v[1];
                w[2] = (__bf16)v[2]; w[3] = (__bf16)v[3];
                *(v4bf*)&Bs[n * BKP + k4] = w;
            }
        }
        __syncthreads();

        // ---- A fragment (ISA 7.12.2 16-bit A 16x32 layout):
        //   lanes 0-15 : row l, K = {0..7, 16..23}
        //   lanes 16-31: row l, K = {8..15, 24..31}
        const int arow = 16 * wave + l;
        v8bf a0 = *(const v8bf*)&As[arow * BKP + half * 8];
        v8bf a1 = *(const v8bf*)&As[arow * BKP + 16 + half * 8];
        v16bf a = __builtin_shufflevector(a0, a1,
                    0, 1, 2, 3, 4, 5, 6, 7, 8, 9, 10, 11, 12, 13, 14, 15);

        // ---- 4 B fragments, 4 WMMAs (B 32x16: lanes 0-15 K={0..7,16..23}, 16-31 K={8..15,24..31})
#pragma unroll
        for (int j = 0; j < 4; ++j) {
            const int brow = 16 * j + l;
            v8bf b0 = *(const v8bf*)&Bs[brow * BKP + half * 8];
            v8bf b1 = *(const v8bf*)&Bs[brow * BKP + 16 + half * 8];
            v16bf b = __builtin_shufflevector(b0, b1,
                        0, 1, 2, 3, 4, 5, 6, 7, 8, 9, 10, 11, 12, 13, 14, 15);
            acc[j] = __builtin_amdgcn_wmma_f32_16x16x32_bf16(
                false, a, false, b, (short)0, acc[j], false, false);
        }
        __syncthreads();
    }

    // ---- epilogue: C/D layout VGPR r -> M = r + 8*half, N = lane&15 ----
#pragma unroll
    for (int j = 0; j < 4; ++j) {
        int n = n0 + 16 * j + l;
        if (n < N) {
#pragma unroll
            for (int r = 0; r < 8; ++r) {
                int mr = m0 + 16 * wave + r + half * 8;
                if (mr < M) {
                    float v = acc[j][r];
                    if (rowScale) v *= rowScale[mr];
                    if (act == 1)      v = fmaxf(v, 0.0f);
                    else if (act == 2) v = 1.0f / (1.0f + __expf(-v));
                    C[(size_t)mr * ldc + n] = v;
                }
            }
        }
    }
}

// s[row] = rsqrt(safe(sum_j A[row, j])), one 256-thread block per row
__global__ void rsqrt_rowsum_kernel(const float* __restrict__ A, float* __restrict__ s,
                                    int N, int lda)
{
    __shared__ float red[256];
    const int row = blockIdx.x;
    const int tid = threadIdx.x;
    const float* a = A + (size_t)row * lda;
    float acc = 0.0f;
    for (int j = tid; j < N; j += 256) acc += a[j];
    red[tid] = acc;
    __syncthreads();
    for (int st = 128; st > 0; st >>= 1) {
        if (tid < st) red[tid] += red[tid + st];
        __syncthreads();
    }
    if (tid == 0) {
        float nsum = red[0];
        if (nsum == 0.0f) nsum = 1.0f;   // safe(): n + (n==0)
        s[row] = rsqrtf(nsum);
    }
}

// s[col] = rsqrt(safe(sum_i A[i, col]))
__global__ void rsqrt_colsum_kernel(const float* __restrict__ A, float* __restrict__ s,
                                    int M, int N, int lda)
{
    int j = blockIdx.x * blockDim.x + threadIdx.x;
    if (j >= N) return;
    float acc = 0.0f;
    for (int i = 0; i < M; ++i) acc += A[(size_t)i * lda + j];
    if (acc == 0.0f) acc = 1.0f;
    s[j] = rsqrtf(acc);
}

// Y[i, j] = (s ? s[i] : 1) * X[i, j]   (also used as strided row copy)
__global__ void scale_rows_kernel(const float* __restrict__ X, const float* __restrict__ s,
                                  float* __restrict__ Y, int M, int N, int ldx, int ldy)
{
    int idx = blockIdx.x * blockDim.x + threadIdx.x;
    if (idx >= M * N) return;
    int i = idx / N, j = idx - i * N;
    float v = X[(size_t)i * ldx + j];
    if (s) v *= s[i];
    Y[(size_t)i * ldy + j] = v;
}

// Per-row top-k: keep the k largest entries (stable tie-break: smaller index),
// zero the rest. One 256-thread block per row; row cached in LDS.
__global__ void topk_rows_kernel(const float* __restrict__ S, float* __restrict__ O,
                                 int cols, int k)
{
    __shared__ float sv[2048];
    __shared__ float redv[256];
    __shared__ int   redi[256];
    const int row = blockIdx.x;
    const int tid = threadIdx.x;
    const float* src = S + (size_t)row * cols;
    float*       dst = O + (size_t)row * cols;

    for (int j = tid; j < cols; j += 256) { sv[j] = src[j]; dst[j] = 0.0f; }
    __syncthreads();

    for (int it = 0; it < k; ++it) {
        float best = -INFINITY; int bi = -1;
        for (int j = tid; j < cols; j += 256) {
            float v = sv[j];
            if (v > best) { best = v; bi = j; }
        }
        redv[tid] = best; redi[tid] = bi;
        __syncthreads();
        for (int st = 128; st > 0; st >>= 1) {
            if (tid < st) {
                float vo = redv[tid + st]; int io = redi[tid + st];
                if (io >= 0 && (vo > redv[tid] ||
                               (vo == redv[tid] && (redi[tid] < 0 || io < redi[tid])))) {
                    redv[tid] = vo; redi[tid] = io;
                }
            }
            __syncthreads();
        }
        if (tid == 0 && redi[0] >= 0) {
            dst[redi[0]] = redv[0];
            sv[redi[0]]  = -INFINITY;
        }
        __syncthreads();
    }
}

// ---------------------------------------------------------------------------
// Host-side orchestration
// ---------------------------------------------------------------------------
static inline void gemm(hipStream_t st, const float* A, const float* B, float* C,
                        int M, int N, int K, int lda, int ldb, int ldc,
                        int tA, int tB, const float* rs, int act)
{
    dim3 grid((unsigned)((N + BN - 1) / BN), (unsigned)((M + BM - 1) / BM));
    dim3 blk(128);
    if (!tA && !tB)
        hipLaunchKernelGGL((gemm_wmma_kernel<0, 0>), grid, blk, 0, st,
                           A, B, C, M, N, K, lda, ldb, ldc, rs, act);
    else if (!tA && tB)
        hipLaunchKernelGGL((gemm_wmma_kernel<0, 1>), grid, blk, 0, st,
                           A, B, C, M, N, K, lda, ldb, ldc, rs, act);
    else if (tA && !tB)
        hipLaunchKernelGGL((gemm_wmma_kernel<1, 0>), grid, blk, 0, st,
                           A, B, C, M, N, K, lda, ldb, ldc, rs, act);
    else
        hipLaunchKernelGGL((gemm_wmma_kernel<1, 1>), grid, blk, 0, st,
                           A, B, C, M, N, K, lda, ldb, ldc, rs, act);
}

// One CGC layer: builds concat[H, H_cross, H_self] @ W with fused norm scaling.
static void cgc_level(hipStream_t st,
                      const float* Radj, const float* nd, const float* nt,
                      const float* Dm, const float* Tm,
                      const float* sD, const float* sT,
                      const float* hd_in, const float* ht_in, int F,
                      const float* Wd, const float* Wt,
                      float* hd_out, float* ht_out,
                      float* Cd, float* Ct, float* tmpD, float* tmpT)
{
    const int F3 = 3 * F;
    // H_dt = relu(diag(nd) * (Radj @ (diag(nt) * ht)))  -> Cd[:, F:2F]
    hipLaunchKernelGGL(scale_rows_kernel, dim3((T_NUM * F + 255) / 256), dim3(256), 0, st,
                       ht_in, nt, tmpT, T_NUM, F, F, F);
    gemm(st, Radj, tmpT, Cd + F, D_NUM, F, T_NUM, T_NUM, F, F3, 0, 0, nd, 1);
    // H_td = relu(diag(nt) * (Radj^T @ (diag(nd) * hd))) -> Ct[:, F:2F]
    hipLaunchKernelGGL(scale_rows_kernel, dim3((D_NUM * F + 255) / 256), dim3(256), 0, st,
                       hd_in, nd, tmpD, D_NUM, F, F, F);
    gemm(st, Radj, tmpD, Ct + F, T_NUM, F, D_NUM, T_NUM, F, F3, 1, 0, nt, 1);
    // H_dd = relu(diag(sD) * (D @ (diag(sD) * hd)))      -> Cd[:, 2F:3F]
    hipLaunchKernelGGL(scale_rows_kernel, dim3((D_NUM * F + 255) / 256), dim3(256), 0, st,
                       hd_in, sD, tmpD, D_NUM, F, F, F);
    gemm(st, Dm, tmpD, Cd + 2 * F, D_NUM, F, D_NUM, D_NUM, F, F3, 0, 0, sD, 1);
    // H_tt = relu(diag(sT) * (T @ (diag(sT) * ht)))      -> Ct[:, 2F:3F]
    hipLaunchKernelGGL(scale_rows_kernel, dim3((T_NUM * F + 255) / 256), dim3(256), 0, st,
                       ht_in, sT, tmpT, T_NUM, F, F, F);
    gemm(st, Tm, tmpT, Ct + 2 * F, T_NUM, F, T_NUM, T_NUM, F, F3, 0, 0, sT, 1);
    // identity slices -> Cd[:, 0:F], Ct[:, 0:F]
    hipLaunchKernelGGL(scale_rows_kernel, dim3((D_NUM * F + 255) / 256), dim3(256), 0, st,
                       hd_in, (const float*)nullptr, Cd, D_NUM, F, F, F3);
    hipLaunchKernelGGL(scale_rows_kernel, dim3((T_NUM * F + 255) / 256), dim3(256), 0, st,
                       ht_in, (const float*)nullptr, Ct, T_NUM, F, F, F3);
    // dense projections
    gemm(st, Cd, Wd, hd_out, D_NUM, UNITS, F3, F3, UNITS, UNITS, 0, 0, nullptr, 1);
    gemm(st, Ct, Wt, ht_out, T_NUM, UNITS, F3, F3, UNITS, UNITS, 0, 0, nullptr, 1);
}

extern "C" void kernel_launch(void* const* d_in, const int* in_sizes, int n_in,
                              void* d_out, int out_size, void* d_ws, size_t ws_size,
                              hipStream_t stream)
{
    (void)in_sizes; (void)n_in; (void)out_size; (void)ws_size;
    const float* R    = (const float*)d_in[0];
    const float* Dm   = (const float*)d_in[1];
    const float* Tm   = (const float*)d_in[2];
    const float* H_d  = (const float*)d_in[3];
    const float* H_t  = (const float*)d_in[4];
    const float* W1g[2] = { (const float*)d_in[5], (const float*)d_in[7] };
    const float* W2g[2] = { (const float*)d_in[6], (const float*)d_in[8] };
    const float* Wdl[2] = { (const float*)d_in[9], (const float*)d_in[11] };
    const float* Wtl[2] = { (const float*)d_in[10], (const float*)d_in[12] };
    const float* Wd_out = (const float*)d_in[13];
    const float* Wt_out = (const float*)d_in[14];
    float* Out = (float*)d_out;

    // ---- carve workspace (floats) ----
    float* p = (float*)d_ws;
    float* S    = p; p += (size_t)D_NUM * T_NUM;   // GL sigmoid scores
    float* Rf   = p; p += (size_t)D_NUM * T_NUM;   // top-k pruned relation
    float* HdG  = p; p += (size_t)D_NUM * UNITS;
    float* HtG  = p; p += (size_t)T_NUM * UNITS;
    float* Cd   = p; p += (size_t)D_NUM * 3 * FEAT;
    float* Ct   = p; p += (size_t)T_NUM * 3 * FEAT;
    float* hdA  = p; p += (size_t)D_NUM * FEAT;
    float* hdB  = p; p += (size_t)D_NUM * FEAT;
    float* htA  = p; p += (size_t)T_NUM * FEAT;
    float* htB  = p; p += (size_t)T_NUM * FEAT;
    float* tmpD = p; p += (size_t)D_NUM * FEAT;
    float* tmpT = p; p += (size_t)T_NUM * FEAT;
    float* sD   = p; p += D_NUM;
    float* sT   = p; p += T_NUM;
    float* ndR  = p; p += D_NUM;
    float* ntR  = p; p += T_NUM;
    float* ndF  = p; p += D_NUM;
    float* ntF  = p; p += T_NUM;

    // ---- degree vectors of the fixed graphs (D, T, R) ----
    hipLaunchKernelGGL(rsqrt_rowsum_kernel, dim3(D_NUM), dim3(256), 0, stream, Dm, sD, D_NUM, D_NUM);
    hipLaunchKernelGGL(rsqrt_rowsum_kernel, dim3(T_NUM), dim3(256), 0, stream, Tm, sT, T_NUM, T_NUM);
    hipLaunchKernelGGL(rsqrt_rowsum_kernel, dim3(D_NUM), dim3(256), 0, stream, R, ndR, T_NUM, T_NUM);
    hipLaunchKernelGGL(rsqrt_colsum_kernel, dim3((T_NUM + 255) / 256), dim3(256), 0, stream,
                       R, ntR, D_NUM, T_NUM, T_NUM);

    const float* hd_cur = H_d;
    const float* ht_cur = H_t;
    float* hd_nxt[2] = { hdA, hdB };
    float* ht_nxt[2] = { htA, htB };
    int F = FEAT;

    for (int lvl = 0; lvl < 2; ++lvl) {
        // ---- GL layer: learned relation graph, top-k pruned ----
        gemm(stream, hd_cur, W1g[lvl], HdG, D_NUM, UNITS, F, F, UNITS, UNITS, 0, 0, nullptr, 0);
        gemm(stream, ht_cur, W2g[lvl], HtG, T_NUM, UNITS, F, F, UNITS, UNITS, 0, 0, nullptr, 0);
        gemm(stream, HdG, HtG, S, D_NUM, T_NUM, UNITS, UNITS, UNITS, T_NUM, 0, 1, nullptr, 2);
        hipLaunchKernelGGL(topk_rows_kernel, dim3(D_NUM), dim3(256), 0, stream, S, Rf, T_NUM, TOPK);
        hipLaunchKernelGGL(rsqrt_rowsum_kernel, dim3(D_NUM), dim3(256), 0, stream, Rf, ndF, T_NUM, T_NUM);
        hipLaunchKernelGGL(rsqrt_colsum_kernel, dim3((T_NUM + 255) / 256), dim3(256), 0, stream,
                           Rf, ntF, D_NUM, T_NUM, T_NUM);
        // ---- CGC layer on pruned graph ----
        cgc_level(stream, Rf, ndF, ntF, Dm, Tm, sD, sT,
                  hd_cur, ht_cur, F, Wdl[lvl], Wtl[lvl],
                  hd_nxt[lvl], ht_nxt[lvl], Cd, Ct, tmpD, tmpT);
        hd_cur = hd_nxt[lvl];
        ht_cur = ht_nxt[lvl];
        F = UNITS;
    }

    // ---- output CGC on the original relation graph ----
    cgc_level(stream, R, ndR, ntR, Dm, Tm, sD, sT,
              hd_cur, ht_cur, UNITS, Wd_out, Wt_out,
              hdA, htA, Cd, Ct, tmpD, tmpT);

    // ---- R_pred = sigmoid(hd @ ht^T) ----
    gemm(stream, hdA, htA, Out, D_NUM, T_NUM, UNITS, UNITS, UNITS, T_NUM, 0, 1, nullptr, 2);
}
